// HyperNet_3633542333210
// MI455X (gfx1250) — matrix operand order repaired and checked by reading, not compiled, not run on hardware
//
#include <hip/hip_runtime.h>
#include <math.h>

// Problem constants (match reference)
#define NN   8192
#define TT   32
#define INF_ 256
#define HH   256
#define GCN  256
#define EE   1024
#define NNZC 65536
#define H3   768

typedef __attribute__((ext_vector_type(16))) __bf16 v16bf;
typedef __attribute__((ext_vector_type(8)))  float  v8f;

// ---------------------------------------------------------------------------
// WMMA helpers: v_wmma_f32_16x16x32_bf16, D = A(16x32) * B(32x16) + C(16x16 f32)
// Layouts per CDNA5 ISA 7.12.2:
//   A: lane m=lane&15 (both halves), dword v -> K = (v>>2)*16 + (lane>>4)*8 + (v&3)*2
//   B: lane col=lane&15, dword v -> K = (v>>2)*16 + (lane>>4)*8 + (v&3)*2
//   C: lane col=lane&15, dword v -> M = (lane>>4)*8 + v
// ---------------------------------------------------------------------------
__device__ __forceinline__ v8f hn_wmma(v16bf a, v16bf b, v8f c) {
  return __builtin_amdgcn_wmma_f32_16x16x32_bf16(false, a, false, b, (short)0, c, false, false);
}

// A fragment from f32 row-major A (lda floats per row)
__device__ __forceinline__ v16bf hn_load_a_f32(const float* __restrict__ A, int lda,
                                               int row_base, int k0, int lane) {
  const float* p = A + (size_t)(row_base + (lane & 15)) * lda + k0 + ((lane >> 4) << 3);
  v16bf a;
#pragma unroll
  for (int v = 0; v < 8; ++v) {
    int kb = ((v >> 2) << 4) + ((v & 3) << 1);
    a[2 * v]     = (__bf16)p[kb];
    a[2 * v + 1] = (__bf16)p[kb + 1];
  }
  return a;
}

// A fragment from bf16 row-major A (no cvt on the critical path; 16B chunks -> b128)
__device__ __forceinline__ v16bf hn_load_a_bf16(const __bf16* __restrict__ A, int lda,
                                                int row_base, int k0, int lane) {
  const __bf16* p = A + (size_t)(row_base + (lane & 15)) * lda + k0 + ((lane >> 4) << 3);
  v16bf a;
#pragma unroll
  for (int v = 0; v < 8; ++v) {
    int kb = ((v >> 2) << 4) + ((v & 3) << 1);
    a[2 * v]     = p[kb];
    a[2 * v + 1] = p[kb + 1];
  }
  return a;
}

// B fragment from bf16 "Bt" = B transposed (Nout x K row-major)
__device__ __forceinline__ v16bf hn_load_b_bf16(const __bf16* __restrict__ Bt, int ldb,
                                                int col_base, int k0, int lane) {
  const __bf16* p = Bt + (size_t)(col_base + (lane & 15)) * ldb + k0 + ((lane >> 4) << 3);
  v16bf b;
#pragma unroll
  for (int v = 0; v < 8; ++v) { int kb = ((v >> 2) << 4) + ((v & 3) << 1); b[2 * v] = p[kb]; b[2 * v + 1] = p[kb + 1]; }
  return b;
}

__device__ __forceinline__ float hn_sigmoid(float x) { return 1.0f / (1.0f + __expf(-x)); }

// ---------------------------------------------------------------------------
// Utility kernels
// ---------------------------------------------------------------------------
__global__ __launch_bounds__(256) void hn_fill_f32(float* p, float v, int n) {
  int i = blockIdx.x * 256 + threadIdx.x;
  if (i < n) p[i] = v;
}

__global__ __launch_bounds__(256) void hn_cvt_bf16(const float* __restrict__ s, __bf16* __restrict__ d, int n) {
  int i = blockIdx.x * 256 + threadIdx.x;
  if (i < n) d[i] = (__bf16)s[i];
}

// s[R][C] f32 -> d[C][R] bf16 (transpose; GEMM B operands then load contiguously)
__global__ __launch_bounds__(256) void hn_cvt_bf16_T(const float* __restrict__ s, __bf16* __restrict__ d,
                                                     int R, int C) {
  int i = blockIdx.x * 256 + threadIdx.x;
  if (i < R * C) {
    int r = i / C, c = i % C;
    d[(size_t)c * R + r] = (__bf16)s[i];
  }
}

__global__ __launch_bounds__(256) void hn_bcast_h0(const float* __restrict__ h0, float* __restrict__ h,
                                                   __bf16* __restrict__ hbf) {
  int i = blockIdx.x * 256 + threadIdx.x;  // grid covers NN*HH
  float v = h0[i & (HH - 1)];
  h[i] = v;
  hbf[i] = (__bf16)v;
}

// ---------------------------------------------------------------------------
// Generic GEMM: C[M x Nout] = A[M x K](f32) @ Bt[Nout x K](bf16)^T, one wave = 16x16 tile
// ---------------------------------------------------------------------------
__global__ __launch_bounds__(32) void hn_gemm(const float* __restrict__ A, const __bf16* __restrict__ Bt,
                                              float* __restrict__ C, int K, int ldc) {
  int lane = threadIdx.x;
  int cb = blockIdx.x << 4, rb = blockIdx.y << 4;
  v8f acc = {};
  for (int k = 0; k < K; k += 32) {
    v16bf a = hn_load_a_f32(A, K, rb, k, lane);
    v16bf b = hn_load_b_bf16(Bt, K, cb, k, lane);
    acc = hn_wmma(a, b, acc);
  }
  int n = cb + (lane & 15), mh = (lane >> 4) << 3;
#pragma unroll
  for (int v = 0; v < 8; ++v) C[(size_t)(rb + mh + v) * ldc + n] = acc[v];
}

// ---------------------------------------------------------------------------
// Fused GRU step: hg = [x_t ; h] @ [W_ih ; W_hh]^T, gates, h_new, last-step scatter.
// One wave = 16 rows x 16 hidden cols (r/z/n gate columns c, c+256, c+512).
// Recurrent A operand comes from the bf16 h shadow written by the previous step.
// ---------------------------------------------------------------------------
__global__ __launch_bounds__(32) void hn_gru_step(const float* __restrict__ xt /* x + t*INF_, lda=TT*INF_ */,
                                                  const float* __restrict__ h_in,
                                                  const __bf16* __restrict__ hbf_in,
                                                  float* __restrict__ h_out,
                                                  __bf16* __restrict__ hbf_out,
                                                  const __bf16* __restrict__ wih /* [3H][IN] bf16 */,
                                                  const __bf16* __restrict__ whh /* [3H][H] bf16 */,
                                                  const float* __restrict__ b_ih,
                                                  const float* __restrict__ b_hh,
                                                  const int* __restrict__ slen,
                                                  float* __restrict__ last, int t) {
  int lane = threadIdx.x;
  int cb = blockIdx.x << 4, rb = blockIdx.y << 4;
  v8f axr = {}, axz = {}, axn = {}, ahr = {}, ahz = {}, ahn = {};
  // input contribution (K = IN = 256), A rows stride TT*INF_
  for (int k = 0; k < INF_; k += 32) {
    v16bf a  = hn_load_a_f32(xt, TT * INF_, rb, k, lane);
    v16bf br = hn_load_b_bf16(wih, INF_, cb, k, lane);
    v16bf bz = hn_load_b_bf16(wih, INF_, cb + HH, k, lane);
    v16bf bn = hn_load_b_bf16(wih, INF_, cb + 2 * HH, k, lane);
    axr = hn_wmma(a, br, axr);
    axz = hn_wmma(a, bz, axz);
    axn = hn_wmma(a, bn, axn);
  }
  // recurrent contribution (K = H = 256) from bf16 shadow
  for (int k = 0; k < HH; k += 32) {
    v16bf a  = hn_load_a_bf16(hbf_in, HH, rb, k, lane);
    v16bf br = hn_load_b_bf16(whh, HH, cb, k, lane);
    v16bf bz = hn_load_b_bf16(whh, HH, cb + HH, k, lane);
    v16bf bn = hn_load_b_bf16(whh, HH, cb + 2 * HH, k, lane);
    ahr = hn_wmma(a, br, ahr);
    ahz = hn_wmma(a, bz, ahz);
    ahn = hn_wmma(a, bn, ahn);
  }
  int n = cb + (lane & 15), mh = (lane >> 4) << 3;
  float bir = b_ih[n], biz = b_ih[n + HH], bin = b_ih[n + 2 * HH];
  float bhr = b_hh[n], bhz = b_hh[n + HH], bhn = b_hh[n + 2 * HH];
#pragma unroll
  for (int v = 0; v < 8; ++v) {
    int m = rb + mh + v;
    float r  = hn_sigmoid(axr[v] + bir + ahr[v] + bhr);
    float z  = hn_sigmoid(axz[v] + biz + ahz[v] + bhz);
    float nn = tanhf(axn[v] + bin + r * (ahn[v] + bhn));
    float hp = h_in[(size_t)m * HH + n];   // f32 state for the update path
    float hv = (1.0f - z) * nn + z * hp;
    h_out[(size_t)m * HH + n] = hv;
    hbf_out[(size_t)m * HH + n] = (__bf16)hv;
    if (slen[m] - 1 == t) last[(size_t)m * HH + n] = hv;
  }
}

// ---------------------------------------------------------------------------
// Hypergraph conv scatters (degrees computed once; float atomics into L2-resident buffers)
// ---------------------------------------------------------------------------
__global__ __launch_bounds__(256) void hn_count_deg(const int* __restrict__ node_idx,
                                                    const int* __restrict__ edge_idx,
                                                    float* __restrict__ Dcnt, float* __restrict__ Bcnt) {
  int i = blockIdx.x * 256 + threadIdx.x;
  if (i < NNZC) {
    atomicAdd(&Dcnt[node_idx[i]], 1.0f);
    atomicAdd(&Bcnt[edge_idx[i]], 1.0f);
  }
}

__global__ __launch_bounds__(256) void hn_invert(float* p, int n) {
  int i = blockIdx.x * 256 + threadIdx.x;
  if (i < n) { float v = p[i]; p[i] = (v > 0.0f) ? 1.0f / v : 0.0f; }
}

__global__ __launch_bounds__(256) void hn_scatter_edges(const float* __restrict__ xw,
                                                        const int* __restrict__ node_idx,
                                                        const int* __restrict__ edge_idx,
                                                        float* __restrict__ ebuf) {
  int nz = blockIdx.x;                      // uniform -> scalar index loads
  int node = node_idx[nz], edge = edge_idx[nz];
  atomicAdd(&ebuf[(size_t)edge * HH + threadIdx.x], xw[(size_t)node * HH + threadIdx.x]);
}

__global__ __launch_bounds__(256) void hn_scale_rows(float* __restrict__ ebuf, const float* __restrict__ Binv) {
  ebuf[(size_t)blockIdx.x * HH + threadIdx.x] *= Binv[blockIdx.x];
}

__global__ __launch_bounds__(256) void hn_scatter_nodes(const float* __restrict__ ebuf,
                                                        const int* __restrict__ node_idx,
                                                        const int* __restrict__ edge_idx,
                                                        float* __restrict__ acc) {
  int nz = blockIdx.x;
  int node = node_idx[nz], edge = edge_idx[nz];
  atomicAdd(&acc[(size_t)node * HH + threadIdx.x], ebuf[(size_t)edge * HH + threadIdx.x]);
}

__global__ __launch_bounds__(256) void hn_hgc_final(const float* __restrict__ acc,
                                                    const float* __restrict__ Dinv,
                                                    const float* __restrict__ bias,
                                                    const float* __restrict__ prev,
                                                    float* __restrict__ out) {
  int m = blockIdx.x, c = threadIdx.x;
  size_t i = (size_t)m * HH + c;
  out[i] = acc[i] * Dinv[m] + bias[c] + prev[i];
}

// ---------------------------------------------------------------------------
// Pass 1: deg[i] = sum_j [ (o3_i . o3_j)/d^2 >= phi ] with diag forced to 1.
// Gram tiles via WMMA on the bf16 o3 shadow; shuffle-reduce per row; atomics.
// ---------------------------------------------------------------------------
#define JCHUNK 16
__global__ __launch_bounds__(32) void hn_adj_deg(const __bf16* __restrict__ o3bf,
                                                 const float* __restrict__ phi_p,
                                                 float* __restrict__ deg) {
  int lane = threadIdx.x;
  int rb = blockIdx.y << 4;
  int j0 = blockIdx.x * (JCHUNK * 16);
  float phi = *phi_p;
  float cnt[8] = {0, 0, 0, 0, 0, 0, 0, 0};
  for (int jb = 0; jb < JCHUNK; ++jb) {
    int jbase = j0 + (jb << 4);
    v8f s = {};
    for (int k = 0; k < HH; k += 32) {
      v16bf a = hn_load_a_bf16(o3bf, HH, rb, k, lane);
      v16bf b = hn_load_b_bf16(o3bf, HH, jbase, k, lane);  // Bt = o3 rows = B columns
      s = hn_wmma(a, b, s);
    }
    int n = jbase + (lane & 15), mh = (lane >> 4) << 3;
#pragma unroll
    for (int v = 0; v < 8; ++v) {
      int m = rb + mh + v;
      float aval = (m == n) ? 1.0f : ((s[v] * (1.0f / 65536.0f)) >= phi ? 1.0f : 0.0f);
      cnt[v] += aval;
    }
  }
#pragma unroll
  for (int v = 0; v < 8; ++v) {
    cnt[v] += __shfl_xor(cnt[v], 1);
    cnt[v] += __shfl_xor(cnt[v], 2);
    cnt[v] += __shfl_xor(cnt[v], 4);
    cnt[v] += __shfl_xor(cnt[v], 8);
  }
  if ((lane & 15) == 0) {
    int mh = (lane >> 4) << 3;
#pragma unroll
    for (int v = 0; v < 8; ++v) atomicAdd(&deg[rb + mh + v], cnt[v]);
  }
}

__global__ __launch_bounds__(256) void hn_dis(const float* __restrict__ deg, float* __restrict__ dis, int n) {
  int i = blockIdx.x * 256 + threadIdx.x;
  if (i < n) { float d = deg[i]; dis[i] = (d > 0.0f) ? rsqrtf(d) : 0.0f; }
}

// gsT[c][m] = bf16(dis[m] * g[m][c])  (transposed so it is a Bt operand with K = NN)
__global__ __launch_bounds__(256) void hn_gsT(const float* __restrict__ g, const float* __restrict__ dis,
                                              __bf16* __restrict__ gsT) {
  int m = blockIdx.x, c = threadIdx.x;
  gsT[(size_t)c * NN + m] = (__bf16)(dis[m] * g[(size_t)m * GCN + c]);
}

// ---------------------------------------------------------------------------
// Pass 2: out[i,:] = dis_i * sum_j A_ij * (dis_j * g_j) + gcn_b.
// 4-wave workgroup per 16-row block; j advances 128/iteration:
//   - wave w computes the 16x32 thresholded gram tile for its own 32-wide j
//     sub-slab (16 WMMA) and stores it to LDS in A-operand row-major form,
//   - barrier, then every wave consumes all four LDS A-fragments against its
//     private 64-column slice of gsT (16 WMMA) -> gram work shared 4-way.
// ---------------------------------------------------------------------------
__global__ __launch_bounds__(128) void hn_gcn_pass2(const __bf16* __restrict__ o3bf,
                                                    const __bf16* __restrict__ gsT,
                                                    const float* __restrict__ dis,
                                                    const float* __restrict__ phi_p,
                                                    const float* __restrict__ gcn_b,
                                                    float* __restrict__ out) {
  __shared__ __bf16 sA[16 * 136];              // 16 rows x 128 jloc (pad to 136 halves)
  int lane = threadIdx.x & 31;
  int w    = threadIdx.x >> 5;                 // wave id 0..3
  int rb   = blockIdx.x << 4;                  // output row block
  int cw   = w << 6;                           // this wave's 64-col output slice
  float phi = *phi_p;
  v8f acc0 = {}, acc1 = {}, acc2 = {}, acc3 = {};
  for (int jj = 0; jj < NN; jj += 128) {
    int j = jj + (w << 5);                     // this wave's 32-wide gram sub-slab
    v8f s0 = {}, s1 = {};
    for (int k = 0; k < HH; k += 32) {
      v16bf a  = hn_load_a_bf16(o3bf, HH, rb, k, lane);
      v16bf b0 = hn_load_b_bf16(o3bf, HH, j, k, lane);
      v16bf b1 = hn_load_b_bf16(o3bf, HH, j + 16, k, lane);
      s0 = hn_wmma(a, b0, s0);
      s1 = hn_wmma(a, b1, s1);
    }
    // threshold + deposit A tile into LDS at jloc = (j - jj) + n
    {
      int n = lane & 15, mh = (lane >> 4) << 3;
      int base = (w << 5);
#pragma unroll
      for (int v = 0; v < 8; ++v) {
        int m = rb + mh + v;
        float a0 = (m == j + n)      ? 1.0f : ((s0[v] * (1.0f / 65536.0f)) >= phi ? 1.0f : 0.0f);
        float a1 = (m == j + 16 + n) ? 1.0f : ((s1[v] * (1.0f / 65536.0f)) >= phi ? 1.0f : 0.0f);
        sA[(mh + v) * 136 + base + n]      = (__bf16)a0;
        sA[(mh + v) * 136 + base + 16 + n] = (__bf16)a1;
      }
    }
    __syncthreads();
    // consume all four 16x32 A-fragments against this wave's gsT column slice
#pragma unroll
    for (int s = 0; s < 4; ++s) {
      v16bf af;
      {
        int row = lane & 15, kh = (lane >> 4) << 3;
#pragma unroll
        for (int v = 0; v < 8; ++v) {
          int kb = ((v >> 2) << 4) + ((v & 3) << 1) + kh;
          af[2 * v]     = sA[row * 136 + (s << 5) + kb];
          af[2 * v + 1] = sA[row * 136 + (s << 5) + kb + 1];
        }
      }
      int k0 = jj + (s << 5);
      v16bf b0 = hn_load_b_bf16(gsT, NN, cw,      k0, lane);
      v16bf b1 = hn_load_b_bf16(gsT, NN, cw + 16, k0, lane);
      v16bf b2 = hn_load_b_bf16(gsT, NN, cw + 32, k0, lane);
      v16bf b3 = hn_load_b_bf16(gsT, NN, cw + 48, k0, lane);
      acc0 = hn_wmma(af, b0, acc0);
      acc1 = hn_wmma(af, b1, acc1);
      acc2 = hn_wmma(af, b2, acc2);
      acc3 = hn_wmma(af, b3, acc3);
    }
    __syncthreads();  // WAR: other waves must finish reading sA before next overwrite
  }
  int n = lane & 15, mh = (lane >> 4) << 3;
#pragma unroll
  for (int v = 0; v < 8; ++v) {
    int m = rb + mh + v;
    float dm = dis[m];
    int c0 = cw + n;
    out[(size_t)m * GCN + c0]      = dm * acc0[v] + gcn_b[c0];
    out[(size_t)m * GCN + c0 + 16] = dm * acc1[v] + gcn_b[c0 + 16];
    out[(size_t)m * GCN + c0 + 32] = dm * acc2[v] + gcn_b[c0 + 32];
    out[(size_t)m * GCN + c0 + 48] = dm * acc3[v] + gcn_b[c0 + 48];
  }
}

// ---------------------------------------------------------------------------
// Host-side orchestration
// ---------------------------------------------------------------------------
extern "C" void kernel_launch(void* const* d_in, const int* in_sizes, int n_in,
                              void* d_out, int out_size, void* d_ws, size_t ws_size,
                              hipStream_t stream) {
  (void)in_sizes; (void)n_in; (void)out_size; (void)ws_size;
  const float* x      = (const float*)d_in[0];
  const int*   hei    = (const int*)d_in[1];
  const int*   node_i = hei;
  const int*   edge_i = hei + NNZC;
  const int*   slen   = (const int*)d_in[2];
  const float* h0     = (const float*)d_in[3];
  const float* W_ih   = (const float*)d_in[4];
  const float* W_hh   = (const float*)d_in[5];
  const float* b_ih   = (const float*)d_in[6];
  const float* b_hh   = (const float*)d_in[7];
  const float* hgc_w[3] = {(const float*)d_in[8], (const float*)d_in[10], (const float*)d_in[12]};
  const float* hgc_b[3] = {(const float*)d_in[9], (const float*)d_in[11], (const float*)d_in[13]};
  const float* phi    = (const float*)d_in[14];
  const float* gcn_w  = (const float*)d_in[15];
  const float* gcn_b  = (const float*)d_in[16];
  float* out = (float*)d_out;

  // Workspace bump allocator (deterministic layout every call)
  char* ws = (char*)d_ws;
  size_t off = 0;
  auto alloc = [&](size_t bytes) -> void* {
    void* p = ws + off;
    off = (off + bytes + 255) & ~(size_t)255;
    return p;
  };
  __bf16* wih_bf = (__bf16*)alloc((size_t)H3 * INF_ * 2);
  __bf16* whh_bf = (__bf16*)alloc((size_t)H3 * HH * 2);
  __bf16* hwT[3] = {(__bf16*)alloc((size_t)HH * HH * 2),
                    (__bf16*)alloc((size_t)HH * HH * 2),
                    (__bf16*)alloc((size_t)HH * HH * 2)};
  __bf16* gwT    = (__bf16*)alloc((size_t)HH * GCN * 2);
  float*  hA     = (float*)alloc((size_t)NN * HH * 4);
  float*  hB     = (float*)alloc((size_t)NN * HH * 4);
  __bf16* hbfA   = (__bf16*)alloc((size_t)NN * HH * 2);
  __bf16* hbfB   = (__bf16*)alloc((size_t)NN * HH * 2);
  float*  last   = (float*)alloc((size_t)NN * HH * 4);
  float*  xw     = (float*)alloc((size_t)NN * HH * 4);   // also reused as g
  float*  accb   = (float*)alloc((size_t)NN * HH * 4);
  float*  o3     = (float*)alloc((size_t)NN * HH * 4);
  __bf16* o3bf   = (__bf16*)alloc((size_t)NN * HH * 2);
  float*  ebuf   = (float*)alloc((size_t)EE * HH * 4);
  __bf16* gsT    = (__bf16*)alloc((size_t)NN * GCN * 2);
  float*  Dinv   = (float*)alloc((size_t)NN * 4);
  float*  Binv   = (float*)alloc((size_t)EE * 4);
  float*  deg    = (float*)alloc((size_t)NN * 4);
  float*  dis    = (float*)alloc((size_t)NN * 4);
  // Reuse GRU ping-pong buffers for hypergraph outputs o1/o2 (GRU done by then)
  float* o1 = hA;
  float* o2 = hB;

  // --- weight conversion (W_ih/W_hh are already "Bt" = [Nout][K]; hgc/gcn need transpose)
  hn_cvt_bf16<<<(H3 * INF_ + 255) / 256, 256, 0, stream>>>(W_ih, wih_bf, H3 * INF_);
  hn_cvt_bf16<<<(H3 * HH + 255) / 256, 256, 0, stream>>>(W_hh, whh_bf, H3 * HH);
  for (int l = 0; l < 3; ++l)
    hn_cvt_bf16_T<<<(HH * HH + 255) / 256, 256, 0, stream>>>(hgc_w[l], hwT[l], HH, HH);
  hn_cvt_bf16_T<<<(HH * GCN + 255) / 256, 256, 0, stream>>>(gcn_w, gwT, HH, GCN);

  // --- GRU: h ping-pong (f32 state + bf16 GEMM shadow), fused WMMA GEMMs + gates
  hn_bcast_h0<<<NN * HH / 256, 256, 0, stream>>>(h0, hA, hbfA);
  hn_fill_f32<<<NN * HH / 256, 256, 0, stream>>>(last, 0.0f, NN * HH);
  for (int t = 0; t < TT; ++t) {
    const float*  hin   = (t & 1) ? hB : hA;
    float*        hout  = (t & 1) ? hA : hB;
    const __bf16* hbin  = (t & 1) ? hbfB : hbfA;
    __bf16*       hbout = (t & 1) ? hbfA : hbfB;
    hn_gru_step<<<dim3(HH / 16, NN / 16), 32, 0, stream>>>(
        x + (size_t)t * INF_, hin, hbin, hout, hbout,
        wih_bf, whh_bf, b_ih, b_hh, slen, last, t);
  }

  // --- node / hyperedge degrees (shared by all three conv layers)
  hn_fill_f32<<<NN / 256, 256, 0, stream>>>(Dinv, 0.0f, NN);
  hn_fill_f32<<<(EE + 255) / 256, 256, 0, stream>>>(Binv, 0.0f, EE);
  hn_count_deg<<<NNZC / 256, 256, 0, stream>>>(node_i, edge_i, Dinv, Binv);
  hn_invert<<<NN / 256, 256, 0, stream>>>(Dinv, NN);
  hn_invert<<<(EE + 255) / 256, 256, 0, stream>>>(Binv, EE);

  // --- three hypergraph conv layers with residual
  const float* prev = last;
  float* outs[3] = {o1, o2, o3};
  for (int l = 0; l < 3; ++l) {
    hn_gemm<<<dim3(HH / 16, NN / 16), 32, 0, stream>>>(prev, hwT[l], xw, HH, HH);
    hn_fill_f32<<<EE * HH / 256, 256, 0, stream>>>(ebuf, 0.0f, EE * HH);
    hn_fill_f32<<<NN * HH / 256, 256, 0, stream>>>(accb, 0.0f, NN * HH);
    hn_scatter_edges<<<NNZC, 256, 0, stream>>>(xw, node_i, edge_i, ebuf);
    hn_scale_rows<<<EE, 256, 0, stream>>>(ebuf, Binv);
    hn_scatter_nodes<<<NNZC, 256, 0, stream>>>(ebuf, node_i, edge_i, accb);
    hn_hgc_final<<<NN, 256, 0, stream>>>(accb, Dinv, hgc_b[l], prev, outs[l]);
    prev = outs[l];
  }

  // --- dynamic adjacency + GCN (bf16 shadow of o3 feeds both gram passes)
  hn_cvt_bf16<<<NN * HH / 256, 256, 0, stream>>>(o3, o3bf, NN * HH);
  float* g = xw;  // reuse
  hn_gemm<<<dim3(GCN / 16, NN / 16), 32, 0, stream>>>(o3, gwT, g, HH, GCN);
  hn_fill_f32<<<NN / 256, 256, 0, stream>>>(deg, 0.0f, NN);
  hn_adj_deg<<<dim3(NN / (JCHUNK * 16), NN / 16), 32, 0, stream>>>(o3bf, phi, deg);
  hn_dis<<<NN / 256, 256, 0, stream>>>(deg, dis, NN);
  hn_gsT<<<NN, GCN, 0, stream>>>(g, dis, gsT);
  hn_gcn_pass2<<<NN / 16, 128, 0, stream>>>(o3bf, gsT, dis, phi, gcn_b, out);
}
